// GenAtlasConditionedOnAge_26388279067316
// MI455X (gfx1250) — matrix operand order repaired and checked by reading, not compiled, not run on hardware
//
#include <hip/hip_runtime.h>

typedef __attribute__((ext_vector_type(16))) _Float16 v16h;
typedef __attribute__((ext_vector_type(8)))  _Float16 v8h;
typedef __attribute__((ext_vector_type(8)))  float    v8f;

#define N1 10242
#define N2 40962
#define N3 163842
#define E1 (3*N1-6)   // 30720
#define E2 (3*N2-6)   // 122880

// ---------------------------------------------------------------- stage 0: hidden = age*W + b (64)
__global__ void hidden_k(const float* __restrict__ age, const float* __restrict__ w,
                         const float* __restrict__ b, float* __restrict__ h) {
  int t = threadIdx.x;
  if (t < 64) h[t] = age[0] * w[t] + b[t];
}

// ---------------------------------------------------------------- stage 1: fc matvec (M x 64) @ h
__global__ void fc_k(const float* __restrict__ h, const float* __restrict__ w,
                     const float* __restrict__ b, float* __restrict__ out, int M) {
  int i = blockIdx.x * 256 + threadIdx.x;
  if (i >= M) return;
  const float4* wr = (const float4*)(w + (size_t)i * 64);
  const float4* hv = (const float4*)h;
  float acc = b[i];
#pragma unroll
  for (int k = 0; k < 16; ++k) {
    float4 a = wr[k], x = hv[k];
    acc += a.x * x.x + a.y * x.y + a.z * x.z + a.w * x.w;
  }
  out[i] = acc;
}

// ---------------------------------------------------------------- BN stats: deterministic 2-stage
template <int C>
__global__ void stats_partial(const float* __restrict__ x, int N, float* __restrict__ partials) {
  __shared__ float sm[256 * 2 * C];
  int tid = threadIdx.x;
  int i = blockIdx.x * 256 + tid;
  float v[C];
#pragma unroll
  for (int c = 0; c < C; ++c) v[c] = (i < N) ? x[(size_t)i * C + c] : 0.f;
#pragma unroll
  for (int c = 0; c < C; ++c) {
    sm[tid * 2 * C + c]     = v[c];
    sm[tid * 2 * C + C + c] = v[c] * v[c];
  }
  __syncthreads();
  for (int s = 128; s > 0; s >>= 1) {
    if (tid < s)
#pragma unroll
      for (int j = 0; j < 2 * C; ++j) sm[tid * 2 * C + j] += sm[(tid + s) * 2 * C + j];
    __syncthreads();
  }
  if (tid < 2 * C) partials[blockIdx.x * 2 * C + tid] = sm[tid];
}

template <int C>
__global__ void stats_finalize(const float* __restrict__ partials, int G, float invN,
                               const float* __restrict__ g, const float* __restrict__ b,
                               float* __restrict__ scsh) {
  __shared__ float sm[256 * 2 * C];
  int tid = threadIdx.x;
  float acc[2 * C];
#pragma unroll
  for (int j = 0; j < 2 * C; ++j) acc[j] = 0.f;
  for (int p = tid; p < G; p += 256)
#pragma unroll
    for (int j = 0; j < 2 * C; ++j) acc[j] += partials[p * 2 * C + j];
#pragma unroll
  for (int j = 0; j < 2 * C; ++j) sm[tid * 2 * C + j] = acc[j];
  __syncthreads();
  for (int s = 128; s > 0; s >>= 1) {
    if (tid < s)
#pragma unroll
      for (int j = 0; j < 2 * C; ++j) sm[tid * 2 * C + j] += sm[(tid + s) * 2 * C + j];
    __syncthreads();
  }
  if (tid < C) {
    float mean = sm[tid] * invN;
    float var  = sm[C + tid] * invN - mean * mean;
    float sc   = g[tid] * rsqrtf(var + 1e-5f);
    scsh[tid]      = sc;
    scsh[16 + tid] = b[tid] - mean * sc;
  }
}

// ---------------------------------------------------------------- fused BN + LeakyReLU + 3->21 linear
__global__ void bn_lrelu_lin21(const float* __restrict__ x, const float* __restrict__ scsh,
                               const float* __restrict__ w, const float* __restrict__ bias,
                               float* __restrict__ ylin, int N) {
  int i = blockIdx.x * 256 + threadIdx.x;
  if (i >= N) return;
  float y[3];
#pragma unroll
  for (int c = 0; c < 3; ++c) {
    float t = x[(size_t)i * 3 + c] * scsh[c] + scsh[16 + c];
    y[c] = t >= 0.f ? t : 0.2f * t;
  }
#pragma unroll
  for (int r = 0; r < 21; ++r)
    ylin[(size_t)i * 21 + r] = bias[r] + w[r * 3 + 0] * y[0] + w[r * 3 + 1] * y[1] + w[r * 3 + 2] * y[2];
}

// ---------------------------------------------------------------- upconv scatter (exact reshape semantics)
__global__ void upconv_scatter(const float* __restrict__ ylin, const int* __restrict__ top,
                               const int* __restrict__ down, float* __restrict__ out,
                               int Nc, int E) {
  int i = blockIdx.x * 256 + threadIdx.x;
  if (i >= Nc + E) return;
  if (i < Nc) {
    int s = top[i];
    out[(size_t)i * 3 + 0] = ylin[(size_t)s * 3 + 0];
    out[(size_t)i * 3 + 1] = ylin[(size_t)s * 3 + 1];
    out[(size_t)i * 3 + 2] = ylin[(size_t)s * 3 + 2];
  } else {
    int e = i - Nc;
#pragma unroll
    for (int cc = 0; cc < 3; ++cc) {
      int f0 = e * 6 + cc * 2;
      int f1 = f0 + 1;
      float a = ylin[(size_t)down[f0 / 3] * 3 + (f0 % 3)];
      float b = ylin[(size_t)down[f1 / 3] * 3 + (f1 % 3)];
      out[(size_t)i * 3 + cc] = 0.5f * (a + b);
    }
  }
}

// ---------------------------------------------------------------- BN + LeakyReLU -> f16, channel-padded
template <int C, int CP>
__global__ void bn_lrelu_tohalf(const float* __restrict__ x, const float* __restrict__ scsh,
                                _Float16* __restrict__ xh, int N) {
  int i = blockIdx.x * 256 + threadIdx.x;
  if (i >= N) return;
#pragma unroll
  for (int c = 0; c < CP; ++c) {
    float t = 0.f;
    if (c < C) {
      t = x[(size_t)i * C + c] * scsh[c] + scsh[16 + c];
      t = t >= 0.f ? t : 0.2f * t;
    }
    xh[(size_t)i * CP + c] = (_Float16)t;
  }
}

// ---------------------------------------------------------------- build weights in WMMA B-fragment layout
// B element mapping (v_wmma_f32_16x16x32_f16): lane l: n=l&15, hi=l>>4; element e -> K = ch*32 + (e>>3)*16 + hi*8 + (e&7).
// Padded-K space: K -> neighbor j = K/CINP, channel c = K%CINP; real weight w[n*(7*CIN) + j*CIN + c].
template <int CIN, int CINP, int COUT>
__global__ void wprep(const float* __restrict__ w, _Float16* __restrict__ wfrag) {
  constexpr int KTOT  = 7 * CINP;
  constexpr int KPAD  = (KTOT + 31) & ~31;
  constexpr int NCHNK = KPAD / 32;
  constexpr int TOTAL = NCHNK * 32 * 16;
  for (int t = threadIdx.x; t < TOTAL; t += 256) {
    int e    = t & 15;
    int lane = (t >> 4) & 31;
    int ch   = t >> 9;
    int hi = lane >> 4, n = lane & 15;
    int K = ch * 32 + (e >> 3) * 16 + hi * 8 + (e & 7);
    int j = K / CINP, c = K % CINP;
    float val = 0.f;
    if (n < COUT && j < 7 && c < CIN) val = w[n * (7 * CIN) + j * CIN + c];
    wfrag[t] = (_Float16)val;
  }
}

// ---------------------------------------------------------------- one-ring conv via v_wmma_f32_16x16x32_f16
// wave = 16-vertex tile (M=16); Cout zero-padded to N=16; K = 7*CINP padded to 32/64.
// Neighbor gather goes straight to LDS with global_load_async_to_lds (ASYNCcnt), then
// A fragments are two ds_load_b128 per K-chunk; B fragments are one 32B global load.
template <int CINP, int COUT>
__global__ void onering_wmma(const _Float16* __restrict__ xh, const int* __restrict__ neigh,
                             const _Float16* __restrict__ wfrag, const float* __restrict__ bias,
                             float* __restrict__ out, int N) {
  constexpr int KTOT  = 7 * CINP;
  constexpr int KPAD  = (KTOT + 31) & ~31;
  constexpr int NCHNK = KPAD / 32;
  __shared__ alignas(16) _Float16 tileA[8][16 * KPAD];

  int wv   = threadIdx.x >> 5;
  int lane = threadIdx.x & 31;
  int tile = blockIdx.x * 8 + wv;
  int v0   = tile * 16;

  _Float16* lds = &tileA[wv][0];

  // zero the K-pad columns (k in [KTOT, KPAD)); KPAD-KTOT is 4 or 8 -> uniform trip count
  constexpr int PADW = KPAD - KTOT;
  for (int idx = lane; idx < 16 * PADW; idx += 32) {
    int row = idx / PADW;
    int k   = KTOT + (idx & (PADW - 1));
    lds[row * KPAD + k] = (_Float16)0.f;
  }

  // async neighbor gather: 16 rows x 7 neighbors, CINP halfs (8 or 16 bytes) each
  for (int t = lane; t < 112; t += 32) {
    int row = t / 7;
    int j   = t - row * 7;
    int v   = v0 + row;
    if (v >= N) v = N - 1;
    int nb = neigh[(size_t)v * 7 + j];
    unsigned           ldsOff = (unsigned)(size_t)(lds + row * KPAD + j * CINP);
    unsigned long long ga     = (unsigned long long)(size_t)(xh + (size_t)nb * CINP);
    if constexpr (CINP == 8) {
      asm volatile("global_load_async_to_lds_b128 %0, %1, off" :: "v"(ldsOff), "v"(ga) : "memory");
    } else {
      asm volatile("global_load_async_to_lds_b64 %0, %1, off" :: "v"(ldsOff), "v"(ga) : "memory");
    }
  }
  asm volatile("s_wait_asynccnt 0x0" ::: "memory");

  int m  = lane & 15;   // A row (lanes 0-15 / 16-31), also D column index
  int hi = lane >> 4;

  v8f cacc = {};
#pragma unroll
  for (int ch = 0; ch < NCHNK; ++ch) {
    // A (16x32 f16): element e -> K = ch*32 + (e>>3)*16 + hi*8 + (e&7): two contiguous 8-half runs
    v8h a0 = *(const v8h*)(lds + m * KPAD + ch * 32 + hi * 8);
    v8h a1 = *(const v8h*)(lds + m * KPAD + ch * 32 + 16 + hi * 8);
    v16h a = __builtin_shufflevector(a0, a1, 0, 1, 2, 3, 4, 5, 6, 7, 8, 9, 10, 11, 12, 13, 14, 15);
    // B: pre-swizzled fragment, 16 contiguous halfs per lane per chunk
    v16h bm = *(const v16h*)(wfrag + (size_t)((ch * 32 + lane) << 4));
    cacc = __builtin_amdgcn_wmma_f32_16x16x32_f16(false, a, false, bm, (short)0, cacc, false, false);
  }

  float bi = (m < COUT) ? bias[m] : 0.f;
  if (v0 + 16 <= N) {
    if (m < COUT) {
#pragma unroll
      for (int r = 0; r < 8; ++r)
        out[(size_t)(v0 + hi * 8 + r) * COUT + m] = cacc[r] + bi;
    }
  } else {
#pragma unroll
    for (int r = 0; r < 8; ++r) {
      int v = v0 + hi * 8 + r;
      if (v < N && m < COUT) out[(size_t)v * COUT + m] = cacc[r] + bi;
    }
  }
}

// ---------------------------------------------------------------- host orchestration
extern "C" void kernel_launch(void* const* d_in, const int* in_sizes, int n_in,
                              void* d_out, int out_size, void* d_ws, size_t ws_size,
                              hipStream_t stream) {
  (void)in_sizes; (void)n_in; (void)out_size; (void)ws_size;

  const float* age      = (const float*)d_in[0];
  const float* fc_age_w = (const float*)d_in[1];
  const float* fc_age_b = (const float*)d_in[2];
  const float* fc_w     = (const float*)d_in[3];
  const float* fc_b     = (const float*)d_in[4];
  const float* bn_up0_g = (const float*)d_in[5];
  const float* bn_up0_b = (const float*)d_in[6];
  const float* up0_w    = (const float*)d_in[7];
  const float* up0_b    = (const float*)d_in[8];
  const float* bn_up1_g = (const float*)d_in[9];
  const float* bn_up1_b = (const float*)d_in[10];
  const float* up1_w    = (const float*)d_in[11];
  const float* up1_b    = (const float*)d_in[12];
  const float* bn0_g    = (const float*)d_in[13];
  const float* bn0_b    = (const float*)d_in[14];
  const float* conv0_w  = (const float*)d_in[15];
  const float* conv0_b  = (const float*)d_in[16];
  const float* bn1_g    = (const float*)d_in[17];
  const float* bn1_b    = (const float*)d_in[18];
  const float* conv1_w  = (const float*)d_in[19];
  const float* conv1_b  = (const float*)d_in[20];
  const float* bn2_g    = (const float*)d_in[21];
  const float* bn2_b    = (const float*)d_in[22];
  const float* conv2_w  = (const float*)d_in[23];
  const float* conv2_b  = (const float*)d_in[24];
  const int*   top0     = (const int*)d_in[25];
  const int*   down0    = (const int*)d_in[26];
  const int*   top1     = (const int*)d_in[27];
  const int*   down1    = (const int*)d_in[28];
  const int*   neigh    = (const int*)d_in[29];

  // workspace layout (float offsets)
  float* ws = (float*)d_ws;
  constexpr size_t OFF_H    = 0;                         // 64
  constexpr size_t OFF_SCSH = 64;                        // 32 (scale[16], shift[16])
  constexpr size_t OFF_PART = 96;                        // 641*16 = 10256
  constexpr size_t OFF_BUFA = 10352;                     // N3*8
  constexpr size_t OFF_BUFB = OFF_BUFA + (size_t)N3 * 8;
  constexpr size_t OFF_YLIN = OFF_BUFB + (size_t)N3 * 8; // N2*21
  constexpr size_t OFF_XH   = (OFF_YLIN + (size_t)N2 * 21 + 7) & ~(size_t)7; // N3*8 halfs
  constexpr size_t OFF_WF   = OFF_XH + (size_t)N3 * 4;   // fragment weights (halfs)

  float* h    = ws + OFF_H;
  float* scsh = ws + OFF_SCSH;
  float* part = ws + OFF_PART;
  float* bufA = ws + OFF_BUFA;
  float* bufB = ws + OFF_BUFB;
  float* ylin = ws + OFF_YLIN;
  _Float16* xh  = (_Float16*)(ws + OFF_XH);
  _Float16* wf0 = (_Float16*)(ws + OFF_WF); // 512 halfs (1 chunk)
  _Float16* wf1 = wf0 + 512;                // 1024 halfs (2 chunks)
  _Float16* wf2 = wf1 + 1024;               // 1024 halfs (2 chunks)

  const int G1 = (N1 + 255) / 256;
  const int G2 = (N2 + 255) / 256;
  const int G3 = (N3 + 255) / 256;
  const int WMMA_BLKS = ((N3 + 15) / 16 + 7) / 8;

  // stage 0/1: age -> hidden -> (N1,3) into bufA
  hipLaunchKernelGGL(hidden_k, dim3(1), dim3(64), 0, stream, age, fc_age_w, fc_age_b, h);
  hipLaunchKernelGGL(fc_k, dim3((3 * N1 + 255) / 256), dim3(256), 0, stream, h, fc_w, fc_b, bufA, 3 * N1);

  // upconv block 0: (N1,3) -> (N2,3) into bufB
  hipLaunchKernelGGL((stats_partial<3>), dim3(G1), dim3(256), 0, stream, bufA, N1, part);
  hipLaunchKernelGGL((stats_finalize<3>), dim3(1), dim3(256), 0, stream, part, G1, 1.0f / N1, bn_up0_g, bn_up0_b, scsh);
  hipLaunchKernelGGL(bn_lrelu_lin21, dim3(G1), dim3(256), 0, stream, bufA, scsh, up0_w, up0_b, ylin, N1);
  hipLaunchKernelGGL(upconv_scatter, dim3(G2), dim3(256), 0, stream, ylin, top0, down0, bufB, N1, E1);

  // upconv block 1: (N2,3) -> (N3,3) into bufA
  hipLaunchKernelGGL((stats_partial<3>), dim3(G2), dim3(256), 0, stream, bufB, N2, part);
  hipLaunchKernelGGL((stats_finalize<3>), dim3(1), dim3(256), 0, stream, part, G2, 1.0f / N2, bn_up1_g, bn_up1_b, scsh);
  hipLaunchKernelGGL(bn_lrelu_lin21, dim3(G2), dim3(256), 0, stream, bufB, scsh, up1_w, up1_b, ylin, N2);
  hipLaunchKernelGGL(upconv_scatter, dim3(G3), dim3(256), 0, stream, ylin, top1, down1, bufA, N2, E2);

  // conv weights -> pre-swizzled f16 B fragments (deterministic, every call)
  hipLaunchKernelGGL((wprep<3, 4, 8>), dim3(1), dim3(256), 0, stream, conv0_w, wf0);
  hipLaunchKernelGGL((wprep<8, 8, 8>), dim3(1), dim3(256), 0, stream, conv1_w, wf1);
  hipLaunchKernelGGL((wprep<8, 8, 2>), dim3(1), dim3(256), 0, stream, conv2_w, wf2);

  // conv0: 3 -> 8 (bufA -> bufB), features padded to 4 halfs/vertex for 8B async transfers
  hipLaunchKernelGGL((stats_partial<3>), dim3(G3), dim3(256), 0, stream, bufA, N3, part);
  hipLaunchKernelGGL((stats_finalize<3>), dim3(1), dim3(256), 0, stream, part, G3, 1.0f / N3, bn0_g, bn0_b, scsh);
  hipLaunchKernelGGL((bn_lrelu_tohalf<3, 4>), dim3(G3), dim3(256), 0, stream, bufA, scsh, xh, N3);
  hipLaunchKernelGGL((onering_wmma<4, 8>), dim3(WMMA_BLKS), dim3(256), 0, stream, xh, neigh, wf0, conv0_b, bufB, N3);

  // conv1: 8 -> 8 (bufB -> bufA)
  hipLaunchKernelGGL((stats_partial<8>), dim3(G3), dim3(256), 0, stream, bufB, N3, part);
  hipLaunchKernelGGL((stats_finalize<8>), dim3(1), dim3(256), 0, stream, part, G3, 1.0f / N3, bn1_g, bn1_b, scsh);
  hipLaunchKernelGGL((bn_lrelu_tohalf<8, 8>), dim3(G3), dim3(256), 0, stream, bufB, scsh, xh, N3);
  hipLaunchKernelGGL((onering_wmma<8, 8>), dim3(WMMA_BLKS), dim3(256), 0, stream, xh, neigh, wf1, conv1_b, bufA, N3);

  // conv2: 8 -> 2 (bufA -> d_out)
  hipLaunchKernelGGL((stats_partial<8>), dim3(G3), dim3(256), 0, stream, bufA, N3, part);
  hipLaunchKernelGGL((stats_finalize<8>), dim3(1), dim3(256), 0, stream, part, G3, 1.0f / N3, bn2_g, bn2_b, scsh);
  hipLaunchKernelGGL((bn_lrelu_tohalf<8, 8>), dim3(G3), dim3(256), 0, stream, bufA, scsh, xh, N3);
  hipLaunchKernelGGL((onering_wmma<8, 2>), dim3(WMMA_BLKS), dim3(256), 0, stream, xh, neigh, wf2, conv2_b, (float*)d_out, N3);
}
